// TemporalExt_55490977464623
// MI455X (gfx1250) — matrix operand order repaired and checked by reading, not compiled, not run on hardware
//
#include <hip/hip_runtime.h>
#include <hip/hip_bf16.h>

typedef __attribute__((ext_vector_type(16))) _Float16 v16h;
typedef __attribute__((ext_vector_type(8)))  _Float16 v8h;
typedef __attribute__((ext_vector_type(8)))  float    v8f;

#define BB     16
#define LL     1024
#define DDIM   512
#define EDIM   1024
#define NSTATE 16
#define DTR    32
#define KCONV  4
#define NVOCAB 128
#define NOUT   256
#define BL     (BB * LL)
#define EPSN   1e-5f

#if defined(__gfx1250__) && __has_builtin(__builtin_amdgcn_global_load_async_to_lds_b32) && \
    __has_builtin(__builtin_amdgcn_s_wait_asynccnt)
#define HAVE_ASYNC_LDS 1
#else
#define HAVE_ASYNC_LDS 0
#endif

#define AS1 __attribute__((address_space(1)))
#define AS3 __attribute__((address_space(3)))

// ---------------------------------------------------------------------------
// f32 -> f16 convert (weights & activations)
// ---------------------------------------------------------------------------
__global__ void k_f32_to_f16(const float* __restrict__ src, _Float16* __restrict__ dst, int n) {
    int i = blockIdx.x * blockDim.x + threadIdx.x;
    if (i < n) dst[i] = (_Float16)src[i];
}

// strided slice convert: dst[r*cols+c] = src[r*src_ld + c]  (dlt columns of dbc)
__global__ void k_slice_to_f16(const float* __restrict__ src, _Float16* __restrict__ dst,
                               int rows, int cols, int src_ld) {
    int i = blockIdx.x * blockDim.x + threadIdx.x;
    if (i >= rows * cols) return;
    int r = i / cols, c = i - r * cols;
    dst[i] = (_Float16)src[(size_t)r * src_ld + c];
}

// ---------------------------------------------------------------------------
// RMSNorm over last dim (d), output f16 for WMMA consumer
// ---------------------------------------------------------------------------
__global__ __launch_bounds__(256)
void k_rmsnorm_f16(const float* __restrict__ x, const float* __restrict__ w,
                   _Float16* __restrict__ out, int d) {
    int row = blockIdx.x;
    const float* xr = x + (size_t)row * d;
    __shared__ float red[256];
    float ss = 0.f;
    for (int i = threadIdx.x; i < d; i += 256) { float v = xr[i]; ss += v * v; }
    red[threadIdx.x] = ss;
    __syncthreads();
    for (int s = 128; s > 0; s >>= 1) {
        if (threadIdx.x < s) red[threadIdx.x] += red[threadIdx.x + s];
        __syncthreads();
    }
    float scale = rsqrtf(red[0] / (float)d + EPSN);
    for (int i = threadIdx.x; i < d; i += 256)
        out[(size_t)row * d + i] = (_Float16)(xr[i] * scale * w[i]);
}

// ---------------------------------------------------------------------------
// WMMA GEMM: C[m,n] = sum_k A[m,k] * W[n,k]  (+bias[n]) (+resid) (softplus)
// A: f16 row-major [M,K] lda.  W: f16 row-major [N,K] ldw.  C: f32 [M,N] ldc.
// One wave computes a 32x64 C tile: 2 A-frags x 4 B-frags, 8 accumulators
// (8 v_wmma per 12 b128 loads -> ~21 f16-FLOPs per byte from L2).
// C/D layout follows CDNA5 ISA 7.12.2 (wave32, f32 C/D); A/B use contiguous K, k = 16*half + s.
// ---------------------------------------------------------------------------
__global__ __launch_bounds__(256)
void k_gemm_wmma(const _Float16* __restrict__ A, int lda,
                 const _Float16* __restrict__ W, int ldw,
                 float* __restrict__ C, int ldc,
                 int M, int N, int K,
                 const float* __restrict__ bias,
                 const float* __restrict__ resid,
                 int act)   // 0 = none, 1 = softplus
{
    int wave   = (int)((blockIdx.x * (unsigned)blockDim.x + threadIdx.x) >> 5);
    int lane   = threadIdx.x & 31;
    int tilesN = N >> 6;                           // 64-wide N tiles
    int tilesM = M >> 5;                           // 32-wide M tiles
    if (wave >= tilesM * tilesN) return;           // wave-uniform: EXEC stays full
    int tm = wave / tilesN;
    int tn = wave - tm * tilesN;

    int lr = lane & 15;       // row-in-tile (A: M row, B: N column)
    int hi = lane >> 4;       // lane-half selects K sub-range

    const _Float16* ap0 = A + (size_t)(tm * 32 + lr) * lda;
    const _Float16* ap1 = ap0 + (size_t)16 * lda;
    const _Float16* bp0 = W + (size_t)(tn * 64 + lr) * ldw;
    const _Float16* bp1 = bp0 + (size_t)16 * ldw;
    const _Float16* bp2 = bp0 + (size_t)32 * ldw;
    const _Float16* bp3 = bp0 + (size_t)48 * ldw;

    v8f acc[2][4];
#pragma unroll
    for (int i = 0; i < 2; ++i)
#pragma unroll
        for (int j = 0; j < 4; ++j) acc[i][j] = (v8f){};

    for (int k0 = 0; k0 < K; k0 += 32) {
        union { v16h v; v8h h[2]; } a0, a1, b0, b1, b2, b3;
        // A 16x32 frag: lane<16 holds K 0..15; lane>=16 holds 16..31
        a0.h[0] = *(const v8h*)(ap0 + k0 + hi * 16);
        a0.h[1] = *(const v8h*)(ap0 + k0 + hi * 16 + 8);
        a1.h[0] = *(const v8h*)(ap1 + k0 + hi * 16);
        a1.h[1] = *(const v8h*)(ap1 + k0 + hi * 16 + 8);
        // B 32x16 frag: lane = column, 16 contiguous K (hi selects K 0..15 / 16..31)
        b0.h[0] = *(const v8h*)(bp0 + k0 + hi * 16);
        b0.h[1] = *(const v8h*)(bp0 + k0 + hi * 16 + 8);
        b1.h[0] = *(const v8h*)(bp1 + k0 + hi * 16);
        b1.h[1] = *(const v8h*)(bp1 + k0 + hi * 16 + 8);
        b2.h[0] = *(const v8h*)(bp2 + k0 + hi * 16);
        b2.h[1] = *(const v8h*)(bp2 + k0 + hi * 16 + 8);
        b3.h[0] = *(const v8h*)(bp3 + k0 + hi * 16);
        b3.h[1] = *(const v8h*)(bp3 + k0 + hi * 16 + 8);

        acc[0][0] = __builtin_amdgcn_wmma_f32_16x16x32_f16(false, a0.v, false, b0.v, (short)0, acc[0][0], false, false);
        acc[0][1] = __builtin_amdgcn_wmma_f32_16x16x32_f16(false, a0.v, false, b1.v, (short)0, acc[0][1], false, false);
        acc[0][2] = __builtin_amdgcn_wmma_f32_16x16x32_f16(false, a0.v, false, b2.v, (short)0, acc[0][2], false, false);
        acc[0][3] = __builtin_amdgcn_wmma_f32_16x16x32_f16(false, a0.v, false, b3.v, (short)0, acc[0][3], false, false);
        acc[1][0] = __builtin_amdgcn_wmma_f32_16x16x32_f16(false, a1.v, false, b0.v, (short)0, acc[1][0], false, false);
        acc[1][1] = __builtin_amdgcn_wmma_f32_16x16x32_f16(false, a1.v, false, b1.v, (short)0, acc[1][1], false, false);
        acc[1][2] = __builtin_amdgcn_wmma_f32_16x16x32_f16(false, a1.v, false, b2.v, (short)0, acc[1][2], false, false);
        acc[1][3] = __builtin_amdgcn_wmma_f32_16x16x32_f16(false, a1.v, false, b3.v, (short)0, acc[1][3], false, false);
    }

    // Epilogue. C layout: lane half selects M base (0/8), VGPR i -> M = base+i, N = lane&15.
#pragma unroll
    for (int j = 0; j < 4; ++j) {
        int n = tn * 64 + j * 16 + lr;
        float bs = bias ? bias[n] : 0.f;
#pragma unroll
        for (int i2 = 0; i2 < 2; ++i2) {
            int mb = tm * 32 + i2 * 16 + hi * 8;
#pragma unroll
            for (int i = 0; i < 8; ++i) {
                float v = acc[i2][j][i] + bs;
                if (resid) v += resid[(size_t)(mb + i) * ldc + n];
                if (act == 1) v = (v > 20.f) ? v : log1pf(__expf(v));
                C[(size_t)(mb + i) * ldc + n] = v;
            }
        }
    }
}

// ---------------------------------------------------------------------------
// Depthwise causal conv1d (K=4, left pad 3) + SiLU, reading the xm half of xz.
// ---------------------------------------------------------------------------
__global__ void k_conv_silu(const float* __restrict__ xz, const float* __restrict__ cw,
                            const float* __restrict__ cb, _Float16* __restrict__ xm16) {
    int i = blockIdx.x * blockDim.x + threadIdx.x;
    if (i >= BB * LL * EDIM) return;
    int e = i % EDIM;
    int l = (i / EDIM) % LL;
    int b = i / (EDIM * LL);
    const float* base = xz + (size_t)(b * LL) * (2 * EDIM) + e;
    float acc = cb[e];
#pragma unroll
    for (int j = 0; j < KCONV; ++j) {
        int t = l - (KCONV - 1) + j;
        if (t >= 0) acc += cw[e * KCONV + j] * base[(size_t)t * (2 * EDIM)];
    }
    float s = acc / (1.f + __expf(-acc));          // silu
    xm16[i] = (_Float16)s;
}

// ---------------------------------------------------------------------------
// Fused selective scan + skip (D*x) + SiLU(z) gate -> y16.
// One block handles (batch b, 256 channels). State h[N=16] in registers.
// B_t/C_t rows are double-buffered in LDS: step t+1's row is fetched with
// global_load_async_to_lds_b32 (ASYNCcnt) while step t computes; per-channel
// delta/x/z operands are software-pipelined one step ahead in registers.
// ---------------------------------------------------------------------------
__global__ __launch_bounds__(256)
void k_scan(const float* __restrict__ delta, const _Float16* __restrict__ xm16,
            const float* __restrict__ dbc, const float* __restrict__ xz,
            const float* __restrict__ A_log, const float* __restrict__ D_param,
            _Float16* __restrict__ y16) {
    const int chunks = EDIM / 256;
    int b = blockIdx.x / chunks;
    int e = (blockIdx.x % chunks) * 256 + threadIdx.x;
    int lane = threadIdx.x;

    float A[NSTATE];
#pragma unroll
    for (int n = 0; n < NSTATE; ++n) A[n] = -__expf(A_log[e * NSTATE + n]);
    float Dv = D_param[e];
    float h[NSTATE];
#pragma unroll
    for (int n = 0; n < NSTATE; ++n) h[n] = 0.f;

    __shared__ float sBC[2][2 * NSTATE];           // [buf][ B(16) | C(16) ]
    const float* bcbase = dbc + (size_t)(b * LL) * (DTR + 2 * NSTATE) + DTR;

    // stage t = 0 into buffer 0
#if HAVE_ASYNC_LDS
    if (lane < 32) {
        __builtin_amdgcn_global_load_async_to_lds_b32(
            (AS1 int*)(bcbase + lane), (AS3 int*)(&sBC[0][lane]), 0, 0);
        __builtin_amdgcn_s_wait_asynccnt(0);
    }
#else
    if (lane < 32) sBC[0][lane] = bcbase[lane];
#endif
    __syncthreads();

    // preload per-channel operands for t = 0
    size_t idx = (size_t)(b * LL) * EDIM + e;
    float d  = delta[idx];
    float xv = (float)xm16[idx];
    float zv = xz[(size_t)(b * LL) * (2 * EDIM) + EDIM + e];

    for (int t = 0; t < LL; ++t) {
        int cur = t & 1, nxt = cur ^ 1;

        // stage B/C row for t+1 into the other buffer (overlaps with compute)
#if HAVE_ASYNC_LDS
        if (t + 1 < LL && lane < 32) {
            __builtin_amdgcn_global_load_async_to_lds_b32(
                (AS1 int*)(bcbase + (size_t)(t + 1) * (DTR + 2 * NSTATE) + lane),
                (AS3 int*)(&sBC[nxt][lane]), 0, 0);
        }
#else
        float stg = 0.f;
        if (t + 1 < LL && lane < 32)
            stg = bcbase[(size_t)(t + 1) * (DTR + 2 * NSTATE) + lane];
#endif
        // prefetch per-channel operands for t+1
        float d2 = 0.f, x2 = 0.f, z2 = 0.f;
        if (t + 1 < LL) {
            size_t idx2 = idx + EDIM;
            d2 = delta[idx2];
            x2 = (float)xm16[idx2];
            z2 = xz[(size_t)(b * LL + t + 1) * (2 * EDIM) + EDIM + e];
        }

        // recurrence step t
        float dx = d * xv, y = 0.f;
#pragma unroll
        for (int n = 0; n < NSTATE; ++n) {
            h[n] = __expf(d * A[n]) * h[n] + dx * sBC[cur][n];
            y += h[n] * sBC[cur][NSTATE + n];
        }
        float yy = y + Dv * xv;
        float sz = zv / (1.f + __expf(-zv));
        y16[idx] = (_Float16)(yy * sz);

        d = d2; xv = x2; zv = z2;
        idx += EDIM;

#if HAVE_ASYNC_LDS
        if (lane < 32) __builtin_amdgcn_s_wait_asynccnt(0);
#else
        if (t + 1 < LL && lane < 32) sBC[nxt][lane] = stg;
#endif
        __syncthreads();
    }
}

// ---------------------------------------------------------------------------
// BatchNorm over seq-len channel: stats over (B, VOCAB) per l
// ---------------------------------------------------------------------------
__global__ __launch_bounds__(256)
void k_bn_stats(const float* __restrict__ logits, float* __restrict__ mu, float* __restrict__ var) {
    int l = blockIdx.x;
    __shared__ float s1[256], s2[256];
    float a = 0.f, q = 0.f;
    for (int i = threadIdx.x; i < BB * NVOCAB; i += 256) {
        int bb = i / NVOCAB, v = i - bb * NVOCAB;
        float x = logits[(size_t)(bb * LL + l) * NVOCAB + v];
        a += x; q += x * x;
    }
    s1[threadIdx.x] = a; s2[threadIdx.x] = q;
    __syncthreads();
    for (int s = 128; s > 0; s >>= 1) {
        if (threadIdx.x < s) { s1[threadIdx.x] += s1[threadIdx.x + s]; s2[threadIdx.x] += s2[threadIdx.x + s]; }
        __syncthreads();
    }
    if (threadIdx.x == 0) {
        float m = s1[0] / (float)(BB * NVOCAB);
        mu[l]  = m;
        var[l] = s2[0] / (float)(BB * NVOCAB) - m * m;
    }
}

__global__ void k_bn_apply(const float* __restrict__ logits, const float* __restrict__ mu,
                           const float* __restrict__ var, const float* __restrict__ gamma,
                           const float* __restrict__ beta, _Float16* __restrict__ out16) {
    int i = blockIdx.x * blockDim.x + threadIdx.x;
    if (i >= BL * NVOCAB) return;
    int l = (i / NVOCAB) % LL;
    float x = logits[i];
    float y = (x - mu[l]) * rsqrtf(var[l] + EPSN) * gamma[l] + beta[l];
    out16[i] = (_Float16)y;
}

// ---------------------------------------------------------------------------
// Host launcher
// ---------------------------------------------------------------------------
static inline void launch_gemm(hipStream_t st, const _Float16* A, int lda,
                               const _Float16* W, int ldw, float* C, int ldc,
                               int M, int N, int K,
                               const float* bias, const float* resid, int act) {
    int waves = (M / 32) * (N / 64);
    int blocks = (waves + 7) / 8;
    k_gemm_wmma<<<blocks, 256, 0, st>>>(A, lda, W, ldw, C, ldc, M, N, K, bias, resid, act);
}

static inline void launch_cvt(hipStream_t st, const float* s, _Float16* d, int n) {
    k_f32_to_f16<<<(n + 255) / 256, 256, 0, st>>>(s, d, n);
}

extern "C" void kernel_launch(void* const* d_in, const int* in_sizes, int n_in,
                              void* d_out, int out_size, void* d_ws, size_t ws_size,
                              hipStream_t stream) {
    const float* x        = (const float*)d_in[0];
    const float* in_w     = (const float*)d_in[1];
    const float* conv_w   = (const float*)d_in[2];
    const float* conv_b   = (const float*)d_in[3];
    const float* xproj_w  = (const float*)d_in[4];
    const float* dtproj_w = (const float*)d_in[5];
    const float* dtproj_b = (const float*)d_in[6];
    const float* A_log    = (const float*)d_in[7];
    const float* D_param  = (const float*)d_in[8];
    const float* out_w    = (const float*)d_in[9];
    const float* rms_w    = (const float*)d_in[10];
    const float* normf_w  = (const float*)d_in[11];
    const float* lm_w     = (const float*)d_in[12];
    const float* bn_g     = (const float*)d_in[13];
    const float* bn_b     = (const float*)d_in[14];
    const float* emb_w    = (const float*)d_in[15];
    const float* emb_b    = (const float*)d_in[16];
    float* outp = (float*)d_out;

    char* ws = (char*)d_ws;
    size_t off = 0;
    auto alloc = [&](size_t bytes) -> void* {
        void* p = ws + off;
        off += (bytes + 255) & ~(size_t)255;
        return p;
    };

    // f32 scratch
    float* h1     = (float*)alloc((size_t)BL * DDIM * 4);
    float* h2     = (float*)alloc((size_t)BL * DDIM * 4);
    float* xz     = (float*)alloc((size_t)BL * 2 * EDIM * 4);
    float* dbc    = (float*)alloc((size_t)BL * 64 * 4);
    float* delta  = (float*)alloc((size_t)BL * EDIM * 4);
    float* logits = (float*)alloc((size_t)BL * NVOCAB * 4);
    float* mu     = (float*)alloc((size_t)LL * 4);
    float* var    = (float*)alloc((size_t)LL * 4);
    // f16 scratch
    _Float16* norm16   = (_Float16*)alloc((size_t)BL * DDIM * 2);
    _Float16* xm16     = (_Float16*)alloc((size_t)BL * EDIM * 2);
    _Float16* dlt16    = (_Float16*)alloc((size_t)BL * DTR * 2);
    _Float16* y16      = (_Float16*)alloc((size_t)BL * EDIM * 2);
    _Float16* h16      = (_Float16*)alloc((size_t)BL * DDIM * 2);
    _Float16* logit16  = (_Float16*)alloc((size_t)BL * NVOCAB * 2);
    // f16 weights
    _Float16* in_w16   = (_Float16*)alloc((size_t)2 * 2 * EDIM * DDIM * 2);
    _Float16* xproj16  = (_Float16*)alloc((size_t)2 * 64 * EDIM * 2);
    _Float16* dtproj16 = (_Float16*)alloc((size_t)2 * EDIM * DTR * 2);
    _Float16* out_w16  = (_Float16*)alloc((size_t)2 * DDIM * EDIM * 2);
    _Float16* lm16     = (_Float16*)alloc((size_t)NVOCAB * DDIM * 2);
    _Float16* emb16    = (_Float16*)alloc((size_t)NOUT * NVOCAB * 2);

    // weight conversion (deterministic, every call)
    launch_cvt(stream, in_w,     in_w16,   2 * 2 * EDIM * DDIM);
    launch_cvt(stream, xproj_w,  xproj16,  2 * 64 * EDIM);
    launch_cvt(stream, dtproj_w, dtproj16, 2 * EDIM * DTR);
    launch_cvt(stream, out_w,    out_w16,  2 * DDIM * EDIM);
    launch_cvt(stream, lm_w,     lm16,     NVOCAB * DDIM);
    launch_cvt(stream, emb_w,    emb16,    NOUT * NVOCAB);

    for (int l = 0; l < 2; ++l) {
        const float* hin = (l == 0) ? x : h1;
        float* hout      = (l == 0) ? h1 : h2;

        k_rmsnorm_f16<<<BL, 256, 0, stream>>>(hin, rms_w + (size_t)l * DDIM, norm16, DDIM);

        // xz = norm @ in_proj_w^T   [BL, 2*ED]
        launch_gemm(stream, norm16, DDIM, in_w16 + (size_t)l * 2 * EDIM * DDIM, DDIM,
                    xz, 2 * EDIM, BL, 2 * EDIM, DDIM, nullptr, nullptr, 0);

        // depthwise causal conv + silu -> xm16
        k_conv_silu<<<(BL * EDIM + 255) / 256, 256, 0, stream>>>(
            xz, conv_w + (size_t)l * EDIM * KCONV, conv_b + (size_t)l * EDIM, xm16);

        // dbc = xm @ x_proj_w^T   [BL, 64]
        launch_gemm(stream, xm16, EDIM, xproj16 + (size_t)l * 64 * EDIM, EDIM,
                    dbc, 64, BL, 64, EDIM, nullptr, nullptr, 0);

        // dlt16 = f16(dbc[:, :32])
        k_slice_to_f16<<<(BL * DTR + 255) / 256, 256, 0, stream>>>(dbc, dlt16, BL, DTR, 64);

        // delta = softplus(dlt @ dt_proj_w^T + b)   [BL, ED]
        launch_gemm(stream, dlt16, DTR, dtproj16 + (size_t)l * EDIM * DTR, DTR,
                    delta, EDIM, BL, EDIM, DTR, dtproj_b + (size_t)l * EDIM, nullptr, 1);

        // fused scan + D*x + silu(z) gate -> y16
        k_scan<<<BB * (EDIM / 256), 256, 0, stream>>>(
            delta, xm16, dbc, xz, A_log + (size_t)l * EDIM * NSTATE,
            D_param + (size_t)l * EDIM, y16);

        // hout = hin + y @ out_proj_w^T
        launch_gemm(stream, y16, EDIM, out_w16 + (size_t)l * DDIM * EDIM, EDIM,
                    hout, DDIM, BL, DDIM, EDIM, nullptr, hin, 0);
    }

    // final norm + head
    k_rmsnorm_f16<<<BL, 256, 0, stream>>>(h2, normf_w, h16, DDIM);
    launch_gemm(stream, h16, DDIM, lm16, DDIM, logits, NVOCAB, BL, NVOCAB, DDIM,
                nullptr, nullptr, 0);
    k_bn_stats<<<LL, 256, 0, stream>>>(logits, mu, var);
    k_bn_apply<<<(BL * NVOCAB + 255) / 256, 256, 0, stream>>>(logits, mu, var, bn_g, bn_b, logit16);
    // out = bn @ emb_w^T + emb_b   [BL, 256]
    launch_gemm(stream, logit16, NVOCAB, emb16, NVOCAB, outp, NOUT, BL, NOUT, NVOCAB,
                emb_b, nullptr, 0);

    (void)in_sizes; (void)n_in; (void)out_size; (void)ws_size;
}
